// GraphConvolutionNetwork_11166914970116
// MI455X (gfx1250) — matrix operand order repaired and checked by reading, not compiled, not run on hardware
//
#include <hip/hip_runtime.h>
#include <hip/hip_bf16.h>
#include <math.h>

// ---------------------------------------------------------------------------
// GraphConvolutionNetwork fused pipeline for MI455X (gfx1250, wave32, WMMA)
//   B=2, N=512, C=256, OUT=256, H=4, hd=64
// All GEMMs run on v_wmma_f32_16x16x32_bf16 with operands pre-staged in bf16
// in WMMA-fragment-friendly layouts (contiguous b128 loads per lane).
// ---------------------------------------------------------------------------

#define Bdim 2
#define Nn   512
#define Cc   256
#define OUTc 256
#define Hh   4
#define HD   64
#define BN   (Bdim * Nn)   // 1024

typedef __bf16 bf16_t;
typedef __attribute__((ext_vector_type(16))) __bf16 v16bf;
typedef __attribute__((ext_vector_type(8)))  __bf16 v8bf;
typedef __attribute__((ext_vector_type(8)))  float  v8f;

// ---------------------------------------------------------------------------
// 16x16 output tile GEMM core (one wave).  A row-major bf16 (lda=K),
// B stored TRANSPOSED bf16 (B^T[n][k], ldb=K).  Fully unrolled K loop.
//   A-fragment layout (16-bit A 16x32): lane row = lane&15; VGPRs carry
//     K = half*16 + 0..15               -> two contiguous 16B loads.
//   B-fragment layout (16-bit B 32x16): lane col = lane&15; VGPRs carry
//     K = half*16 + 0..15               -> one contiguous 32B load.
// ---------------------------------------------------------------------------
template <int K>
__device__ __forceinline__ v8f gemm_tile(const bf16_t* __restrict__ Arow,
                                         const bf16_t* __restrict__ BTcol,
                                         int half)
{
    v8f acc = {};
#pragma unroll
    for (int k0 = 0; k0 < K; k0 += 32) {
        const v8bf alo = *(const v8bf*)(Arow + k0 + half * 16);
        const v8bf ahi = *(const v8bf*)(Arow + k0 + half * 16 + 8);
        const v16bf a = __builtin_shufflevector(
            alo, ahi, 0, 1, 2, 3, 4, 5, 6, 7, 8, 9, 10, 11, 12, 13, 14, 15);
        const v16bf b = *(const v16bf*)(BTcol + k0 + half * 16);
        acc = __builtin_amdgcn_wmma_f32_16x16x32_bf16(
            false, a, false, b, (short)0, acc, false, false);
    }
    return acc;
}

// ---------------------------------------------------------------------------
// Prep 1: x -> bf16 (row-major, A operand).  4 elems/thread.
// ---------------------------------------------------------------------------
__global__ __launch_bounds__(256) void cvt_x_kernel(
    const float* __restrict__ x, bf16_t* __restrict__ xbf)
{
    const int idx = (blockIdx.x * 256 + threadIdx.x) * 4;
    const float4 v = *(const float4*)(x + idx);
    xbf[idx + 0] = (__bf16)v.x;
    xbf[idx + 1] = (__bf16)v.y;
    xbf[idx + 2] = (__bf16)v.z;
    xbf[idx + 3] = (__bf16)v.w;
}

// ---------------------------------------------------------------------------
// Prep 2: four 256x256 weight matrices -> bf16 TRANSPOSED (B operand).
// blockIdx.y selects matrix (uniform); block = output row n, thread = k.
// ---------------------------------------------------------------------------
__global__ __launch_bounds__(256) void cvt_wT_kernel(
    const float* __restrict__ s0, const float* __restrict__ s1,
    const float* __restrict__ s2, const float* __restrict__ s3,
    bf16_t* __restrict__ d0, bf16_t* __restrict__ d1,
    bf16_t* __restrict__ d2, bf16_t* __restrict__ d3)
{
    const int n = blockIdx.x;
    const int k = threadIdx.x;
    const float* s;
    bf16_t* d;
    switch (blockIdx.y) {
        case 0:  s = s0; d = d0; break;
        case 1:  s = s1; d = d1; break;
        case 2:  s = s2; d = d2; break;
        default: s = s3; d = d3; break;
    }
    d[(size_t)n * 256 + k] = (__bf16)s[(size_t)k * 256 + n];
}

// ---------------------------------------------------------------------------
// Kernel: generic [1024x256] = Abf[1024xK=256] @ B (+bias) -> f32 out.
// Used for Hi (bias=be) and Hj (bias=null).  1024 tiles, 4 waves/blk.
// ---------------------------------------------------------------------------
__global__ __launch_bounds__(128) void proj_f32_kernel(
    const bf16_t* __restrict__ Abf, const bf16_t* __restrict__ BT,
    const float* __restrict__ bias, float* __restrict__ out)
{
    const int lane = threadIdx.x & 31;
    const int wave = threadIdx.x >> 5;
    const int tile = blockIdx.x * 4 + wave;        // 0..1023
    const int m0 = (tile >> 4) << 4;
    const int n0 = (tile & 15) << 4;
    const int half = lane >> 4, l15 = lane & 15;
    const int rowA = m0 + l15, colB = n0 + l15;

    const v8f acc = gemm_tile<Cc>(Abf + (size_t)rowA * Cc,
                                  BT + (size_t)colB * Cc, half);
    const float bval = bias ? bias[colB] : 0.0f;
#pragma unroll
    for (int r = 0; r < 8; ++r) {
        const int m = m0 + r + half * 8;
        out[(size_t)m * OUTc + colB] = acc[r] + bval;
    }
}

// ---------------------------------------------------------------------------
// Kernel: V projection -> bf16 stored TRANSPOSED per head: VT[b][h][d][j].
// (so the P@V GEMM gets contiguous B-operand loads)
// ---------------------------------------------------------------------------
__global__ __launch_bounds__(128) void proj_v_kernel(
    const bf16_t* __restrict__ Abf, const bf16_t* __restrict__ WvT,
    const float* __restrict__ bv, bf16_t* __restrict__ VT)
{
    const int lane = threadIdx.x & 31;
    const int wave = threadIdx.x >> 5;
    const int tile = blockIdx.x * 4 + wave;        // 0..1023
    const int m0 = (tile >> 4) << 4;
    const int n0 = (tile & 15) << 4;
    const int half = lane >> 4, l15 = lane & 15;
    const int rowA = m0 + l15, colB = n0 + l15;

    const v8f acc = gemm_tile<Cc>(Abf + (size_t)rowA * Cc,
                                  WvT + (size_t)colB * Cc, half);
    const float bval = bv[colB];
    const int h = colB / HD, d = colB % HD;
#pragma unroll
    for (int r = 0; r < 8; ++r) {
        const int m = m0 + r + half * 8;           // (b, j)
        const int b = m / Nn, j = m % Nn;
        VT[(((size_t)(b * Hh + h)) * HD + d) * Nn + j] = (__bf16)(acc[r] + bval);
    }
}

// ---------------------------------------------------------------------------
// Kernel: edge logits (dominant VALU kernel).
// Wave = (bi, jchunk of 128); 8 channels/lane; LN via wave reductions;
// exact erf-GELU; 4-head dot with W2.  4096 waves total.
// ---------------------------------------------------------------------------
__global__ __launch_bounds__(256) void logits_kernel(
    const float* __restrict__ Hi, const float* __restrict__ Hj,
    const float* __restrict__ ln1_g, const float* __restrict__ ln1_b,
    const float* __restrict__ W2,    // [OUT][H]
    const float* __restrict__ b2,    // [H]
    float* __restrict__ logits)      // [(b*N+i)*H + h]*N + j
{
    const int lane = threadIdx.x & 31;
    const int wave = threadIdx.x >> 5;
    const int wid  = blockIdx.x * 8 + wave;     // 0..4095
    const int bi   = wid >> 2;                  // 0..1023
    const int jc   = wid & 3;                   // j chunk
    const int b    = bi / Nn;
    const int c0   = lane * 8;

    float g[8], bb[8], hi[8], w2l[8][4];
#pragma unroll
    for (int u = 0; u < 8; ++u) {
        g[u]  = ln1_g[c0 + u];
        bb[u] = ln1_b[c0 + u];
        hi[u] = Hi[(size_t)bi * OUTc + c0 + u];
#pragma unroll
        for (int h = 0; h < 4; ++h) w2l[u][h] = W2[(size_t)(c0 + u) * Hh + h];
    }
    const float b2l[4] = { b2[0], b2[1], b2[2], b2[3] };
    const float inv = 1.0f / (float)OUTc;

    const int j0 = jc * (Nn / 4), j1 = j0 + (Nn / 4);
    for (int j = j0; j < j1; ++j) {
        const float* hjrow = Hj + ((size_t)b * Nn + j) * OUTc + c0;
        __builtin_prefetch(hjrow + OUTc, 0, 1);    // next j row
        const float4 h0 = *(const float4*)(hjrow);
        const float4 h1 = *(const float4*)(hjrow + 4);
        float t[8] = { hi[0] + h0.x, hi[1] + h0.y, hi[2] + h0.z, hi[3] + h0.w,
                       hi[4] + h1.x, hi[5] + h1.y, hi[6] + h1.z, hi[7] + h1.w };
        float s = 0.0f, ss = 0.0f;
#pragma unroll
        for (int u = 0; u < 8; ++u) { s += t[u]; ss += t[u] * t[u]; }
#pragma unroll
        for (int m = 16; m >= 1; m >>= 1) {
            s  += __shfl_xor(s,  m, 32);
            ss += __shfl_xor(ss, m, 32);
        }
        const float mu   = s * inv;
        const float rstd = rsqrtf(ss * inv - mu * mu + 1e-5f);

        float p[4] = { 0.0f, 0.0f, 0.0f, 0.0f };
#pragma unroll
        for (int u = 0; u < 8; ++u) {
            const float yn = (t[u] - mu) * rstd * g[u] + bb[u];
            const float ge = 0.5f * yn * (1.0f + erff(yn * 0.70710678118654752f));
#pragma unroll
            for (int h = 0; h < 4; ++h) p[h] += ge * w2l[u][h];
        }
#pragma unroll
        for (int m = 16; m >= 1; m >>= 1) {
#pragma unroll
            for (int h = 0; h < 4; ++h) p[h] += __shfl_xor(p[h], m, 32);
        }
        if (lane < 4)
            logits[((size_t)bi * Hh + lane) * Nn + j] = p[lane] + b2l[lane];
    }
}

// ---------------------------------------------------------------------------
// Kernel: softmax over j -> P bf16 in [b][h][i][j] (WMMA A layout source).
// One wave per (b,i,h) row of 512; 16 elems/lane.
// ---------------------------------------------------------------------------
__global__ __launch_bounds__(256) void softmax_kernel(
    const float* __restrict__ logits, bf16_t* __restrict__ P)
{
    const int lane = threadIdx.x & 31;
    const int wave = threadIdx.x >> 5;
    const int row  = blockIdx.x * 8 + wave;   // 0..4095 == bi*H + h
    const int bi = row / Hh, h = row % Hh;
    const int b  = bi / Nn,  i = bi % Nn;

    const float* src = logits + (size_t)row * Nn;
    float vals[16];
    float mx = -3.4e38f;
#pragma unroll
    for (int u = 0; u < 16; ++u) {
        vals[u] = src[lane + 32 * u];
        mx = fmaxf(mx, vals[u]);
    }
#pragma unroll
    for (int m = 16; m >= 1; m >>= 1) mx = fmaxf(mx, __shfl_xor(mx, m, 32));
    float sum = 0.0f;
#pragma unroll
    for (int u = 0; u < 16; ++u) { vals[u] = __expf(vals[u] - mx); sum += vals[u]; }
#pragma unroll
    for (int m = 16; m >= 1; m >>= 1) sum += __shfl_xor(sum, m, 32);
    const float rs = 1.0f / sum;

    bf16_t* dst = P + (((size_t)(b * Hh + h)) * Nn + i) * Nn;
#pragma unroll
    for (int u = 0; u < 16; ++u) dst[lane + 32 * u] = (__bf16)(vals[u] * rs);
}

// ---------------------------------------------------------------------------
// Kernel: attention aggregation  O[b,h] = P[b,h](512x512) @ V[b,h](512x64).
// A = P rows (bf16), B = VT cols (bf16, contiguous).  Output bf16 row-major
// [BN][OUT] so it feeds the next GEMM's A operand directly.
// ---------------------------------------------------------------------------
__global__ __launch_bounds__(128) void attn_kernel(
    const bf16_t* __restrict__ P,    // [b][h][i][j]
    const bf16_t* __restrict__ VT,   // [b][h][d][j]
    bf16_t* __restrict__ outh)       // [(b*N+i)][h*64+d] bf16
{
    const int lane = threadIdx.x & 31;
    const int wave = threadIdx.x >> 5;
    const int tile = blockIdx.x * 4 + wave;        // 0..1023
    const int bh  = tile >> 7;                     // (b,h): 128 tiles each
    const int rem = tile & 127;
    const int mt  = rem >> 2, nt = rem & 3;
    const int b = bh / Hh, h = bh % Hh;
    const int half = lane >> 4, l15 = lane & 15;

    const int i = mt * 16 + l15;
    const int d = nt * 16 + l15;
    const bf16_t* Arow  = P  + ((size_t)bh * Nn + i) * Nn;
    const bf16_t* BTcol = VT + ((size_t)bh * HD + d) * Nn;

    const v8f acc = gemm_tile<Nn>(Arow, BTcol, half);
#pragma unroll
    for (int r = 0; r < 8; ++r) {
        const int ii = mt * 16 + r + half * 8;
        outh[((size_t)(b * Nn + ii)) * OUTc + h * HD + d] = (__bf16)acc[r];
    }
}

// ---------------------------------------------------------------------------
// Kernel: output projection + residual:  tmp = outh @ Wo + bo + x  (f32)
// ---------------------------------------------------------------------------
__global__ __launch_bounds__(128) void outproj_kernel(
    const bf16_t* __restrict__ outh, const bf16_t* __restrict__ WoT,
    const float* __restrict__ bo, const float* __restrict__ x,
    float* __restrict__ tmp)
{
    const int lane = threadIdx.x & 31;
    const int wave = threadIdx.x >> 5;
    const int tile = blockIdx.x * 4 + wave;        // 0..1023
    const int m0 = (tile >> 4) << 4;
    const int n0 = (tile & 15) << 4;
    const int half = lane >> 4, l15 = lane & 15;
    const int rowA = m0 + l15, colB = n0 + l15;

    const v8f acc = gemm_tile<OUTc>(outh + (size_t)rowA * OUTc,
                                    WoT + (size_t)colB * OUTc, half);
    const float bon = bo[colB];
#pragma unroll
    for (int r = 0; r < 8; ++r) {
        const int m = m0 + r + half * 8;
        const size_t idx = (size_t)m * OUTc + colB;
        tmp[idx] = acc[r] + bon + x[idx];
    }
}

// ---------------------------------------------------------------------------
// Kernel: final LayerNorm.  One wave per row, 8 chans/lane.
// ---------------------------------------------------------------------------
__global__ __launch_bounds__(256) void ln2_kernel(
    const float* __restrict__ tmp,
    const float* __restrict__ g, const float* __restrict__ bb,
    float* __restrict__ out)
{
    const int lane = threadIdx.x & 31;
    const int wave = threadIdx.x >> 5;
    const int row  = blockIdx.x * 8 + wave;     // 0..1023
    const int c0   = lane * 8;
    const float* src = tmp + (size_t)row * OUTc;

    const float4 t0 = *(const float4*)(src + c0);
    const float4 t1 = *(const float4*)(src + c0 + 4);
    float t[8] = { t0.x, t0.y, t0.z, t0.w, t1.x, t1.y, t1.z, t1.w };
    float s = 0.0f, ss = 0.0f;
#pragma unroll
    for (int u = 0; u < 8; ++u) { s += t[u]; ss += t[u] * t[u]; }
#pragma unroll
    for (int m = 16; m >= 1; m >>= 1) {
        s  += __shfl_xor(s,  m, 32);
        ss += __shfl_xor(ss, m, 32);
    }
    const float inv  = 1.0f / (float)OUTc;
    const float mu   = s * inv;
    const float rstd = rsqrtf(ss * inv - mu * mu + 1e-5f);
#pragma unroll
    for (int u = 0; u < 8; ++u)
        out[(size_t)row * OUTc + c0 + u] = (t[u] - mu) * rstd * g[c0 + u] + bb[c0 + u];
}

// ---------------------------------------------------------------------------
// Host launcher.  Workspace layout (bytes):
//   xbf     @ 0          (512 KB)   WeT0 @ 512K  (128 KB)  WeT1 @ 640K (128 KB)
//   WvT     @ 768K       (128 KB)   WoT  @ 896K  (128 KB)
//   Hi      @ 1 MB       (1 MB)     Hj   @ 2 MB  (1 MB)
//   VT      @ 3 MB       (512 KB)   logits @ 3.5 MB (8 MB)
//   P       @ 11.5 MB    (4 MB)     outh @ 15.5 MB (512 KB)
//   tmp     @ 16 MB      (1 MB)     total ~17 MB
// ---------------------------------------------------------------------------
extern "C" void kernel_launch(void* const* d_in, const int* in_sizes, int n_in,
                              void* d_out, int out_size, void* d_ws, size_t ws_size,
                              hipStream_t stream)
{
    (void)in_sizes; (void)n_in; (void)out_size; (void)ws_size;

    const float* x     = (const float*)d_in[0];
    const float* We    = (const float*)d_in[1];
    const float* be    = (const float*)d_in[2];
    const float* ln1_g = (const float*)d_in[3];
    const float* ln1_b = (const float*)d_in[4];
    const float* W2    = (const float*)d_in[5];
    const float* b2    = (const float*)d_in[6];
    const float* Wv    = (const float*)d_in[7];
    const float* bv    = (const float*)d_in[8];
    const float* Wo    = (const float*)d_in[9];
    const float* bo    = (const float*)d_in[10];
    const float* ln2_g = (const float*)d_in[11];
    const float* ln2_b = (const float*)d_in[12];

    char* ws = (char*)d_ws;
    bf16_t* xbf    = (bf16_t*)(ws);
    bf16_t* WeT0   = (bf16_t*)(ws + (size_t)524288);
    bf16_t* WeT1   = (bf16_t*)(ws + (size_t)655360);
    bf16_t* WvT    = (bf16_t*)(ws + (size_t)786432);
    bf16_t* WoT    = (bf16_t*)(ws + (size_t)917504);
    float*  Hi     = (float*)(ws + (size_t)1048576);
    float*  Hj     = (float*)(ws + (size_t)2097152);
    bf16_t* VT     = (bf16_t*)(ws + (size_t)3145728);
    float*  logits = (float*)(ws + (size_t)3670016);
    bf16_t* P      = (bf16_t*)(ws + (size_t)12058624);
    bf16_t* outh   = (bf16_t*)(ws + (size_t)16252928);
    float*  tmp    = (float*)(ws + (size_t)16777216);
    float*  out    = (float*)d_out;

    cvt_x_kernel  <<<256, 256, 0, stream>>>(x, xbf);
    cvt_wT_kernel <<<dim3(256, 4), 256, 0, stream>>>(
        We, We + (size_t)Cc * OUTc, Wv, Wo, WeT0, WeT1, WvT, WoT);

    proj_f32_kernel<<<256, 128, 0, stream>>>(xbf, WeT0, be, Hi);
    proj_f32_kernel<<<256, 128, 0, stream>>>(xbf, WeT1, (const float*)nullptr, Hj);
    proj_v_kernel  <<<256, 128, 0, stream>>>(xbf, WvT, bv, VT);

    logits_kernel <<<512, 256, 0, stream>>>(Hi, Hj, ln1_g, ln1_b, W2, b2, logits);
    softmax_kernel<<<512, 256, 0, stream>>>(logits, P);
    attn_kernel   <<<256, 128, 0, stream>>>(P, VT, outh);
    outproj_kernel<<<256, 128, 0, stream>>>(outh, WoT, bo, x, tmp);
    ln2_kernel    <<<128, 256, 0, stream>>>(tmp, ln2_g, ln2_b, out);
}
